// WeaklySelector_10471130268210
// MI455X (gfx1250) — compile-verified, matrix operand order checked by reading, not run on hardware
//
#include <hip/hip_runtime.h>
#include <hip/hip_bf16.h>

typedef __attribute__((ext_vector_type(16))) _Float16 v16h;
typedef __attribute__((ext_vector_type(8)))  _Float16 v8h;
typedef __attribute__((ext_vector_type(8)))  float    v8f;

#define BATCH   32
#define SEQ     1024
#define CH      1536
#define NCLS    200
#define NTILES  13       // 13 x 16 = 208 columns (200 valid)
#define NSEL    128
#define KC      128      // K-chunk staged in LDS (4 WMMA K-steps)
#define WSTR    136      // LDS row stride in halfs (136*2=272B -> conflict-free)

// Pack 8 fp32 (two float4) into 8 consecutive f16 slots of a v16h fragment.
__device__ __forceinline__ void pack8(v16h& dst, int base, float4 a, float4 b) {
    dst[base + 0] = (_Float16)a.x; dst[base + 1] = (_Float16)a.y;
    dst[base + 2] = (_Float16)a.z; dst[base + 3] = (_Float16)a.w;
    dst[base + 4] = (_Float16)b.x; dst[base + 5] = (_Float16)b.y;
    dst[base + 6] = (_Float16)b.z; dst[base + 7] = (_Float16)b.w;
}

// CDNA5 16-bit A-fragment (16x32): lane<16 holds K klo+0..7 (v0..3) and
// klo+16..23 (v4..7); lane>=16 same with klo=8. rowPtr already includes klo.
__device__ __forceinline__ v16h load_frag_global(const float* __restrict__ rowPtr, int kb) {
    float4 f0 = *(const float4*)(rowPtr + kb + 0);
    float4 f1 = *(const float4*)(rowPtr + kb + 4);
    float4 f2 = *(const float4*)(rowPtr + kb + 16);
    float4 f3 = *(const float4*)(rowPtr + kb + 20);
    v16h r;
    pack8(r, 0, f0, f1);
    pack8(r, 8, f2, f3);
    return r;
}

// Same fragment shape sourced from the f16 W tile staged in LDS (two b128 loads).
__device__ __forceinline__ v16h load_frag_lds(const _Float16* p) {
    v8h lo = *(const v8h*)(p);
    v8h hi = *(const v8h*)(p + 16);
    v16h r;
    #pragma unroll
    for (int i = 0; i < 8; ++i) { r[i] = lo[i]; r[8 + i] = hi[i]; }
    return r;
}

// Kernel 1: block = (batch, 128-row S tile); wave w owns rows [w*16, w*16+16).
// Each wave computes all 208 logit columns for its rows (13 WMMAs per K-step,
// A fragment loaded once from HBM, B fragments double-buffered from the
// LDS-staged W chunk), then reduces conf = 1/sum(exp(logit-max)) in registers.
__global__ __launch_bounds__(256)
void ws_gemm_conf_kernel(const float* __restrict__ x,
                         const float* __restrict__ W,
                         const float* __restrict__ bias,
                         float* __restrict__ conf) {
    __shared__ _Float16 ldsW[NTILES * 16 * WSTR];   // 208 x 136 halfs = 56.5 KB

    const int b     = blockIdx.x >> 3;              // 8 S-tiles of 128 per batch
    const int sBase = (blockIdx.x & 7) * 128;

    const int tid  = threadIdx.x;
    const int lane = tid & 31;
    const int wave = tid >> 5;
    const int half = lane >> 4;                     // 0: K-lo half, 1: K-hi half
    const int l15  = lane & 15;
    const int klo  = half ? 8 : 0;

    // This wave's A rows (read from HBM exactly once).
    const int rowBase = sBase + wave * 16;
    const float* aRow = x + ((size_t)b * SEQ + rowBase + l15) * (size_t)CH + klo;

    v8f acc[NTILES];
    #pragma unroll
    for (int t = 0; t < NTILES; ++t) acc[t] = (v8f){};

    for (int kc = 0; kc < CH; kc += KC) {
        // ---- Cooperative stage: W[0..207][kc..kc+127] -> f16 in LDS ----
        // 8 floats per step: 2 x global_load_b128 -> cvt -> 1 x ds_store_b128.
        #pragma unroll 2
        for (int e = tid; e < NTILES * 16 * (KC / 8); e += 256) {
            const int row = e >> 4;                 // KC/8 = 16 groups per row
            const int c8  = e & 15;
            const int gr  = row < NCLS ? row : NCLS - 1;   // clamp pad rows
            const float* gp = W + (size_t)gr * CH + kc + c8 * 8;
            float4 f0 = *(const float4*)(gp);
            float4 f1 = *(const float4*)(gp + 4);
            v8h h;
            h[0] = (_Float16)f0.x; h[1] = (_Float16)f0.y;
            h[2] = (_Float16)f0.z; h[3] = (_Float16)f0.w;
            h[4] = (_Float16)f1.x; h[5] = (_Float16)f1.y;
            h[6] = (_Float16)f1.z; h[7] = (_Float16)f1.w;
            *(v8h*)(&ldsW[row * WSTR + c8 * 8]) = h;
        }
        __syncthreads();

        // ---- 4 WMMA K-steps over the staged chunk ----
        #pragma unroll
        for (int ks = 0; ks < KC / 32; ++ks) {
            const int kb = kc + ks * 32;
            __builtin_prefetch(aRow + kb + KC, 0, 1);       // global_prefetch_b8
            v16h A = load_frag_global(aRow, kb);
            const _Float16* bBase = &ldsW[l15 * WSTR + ks * 32 + klo];
            // Double-buffered B fragments: issue loads for tile t+1 before
            // the WMMA for tile t so LDS latency hides under matrix issue.
            v16h Bcur = load_frag_lds(bBase);
            #pragma unroll
            for (int t = 0; t < NTILES; ++t) {
                v16h Bnext;
                if (t < NTILES - 1)
                    Bnext = load_frag_lds(bBase + (t + 1) * 16 * WSTR);
                acc[t] = __builtin_amdgcn_wmma_f32_16x16x32_f16(
                             false, A, false, Bcur, (short)0, acc[t], false, false);
                Bcur = Bnext;
            }
        }
        __syncthreads();
    }

    // ---- Bias ----
    #pragma unroll
    for (int t = 0; t < NTILES; ++t) {
        const int col = t * 16 + l15;
        const float bv = bias[col < NCLS ? col : 0];
        #pragma unroll
        for (int v = 0; v < 8; ++v) acc[t][v] += bv;
    }

    // ---- In-register softmax confidence per row ----
    // acc[t][v]: lanes 0-15 -> row v, lanes 16-31 -> row v+8; col = t*16+l15.
    const bool lastValid = (12 * 16 + l15) < NCLS;  // tile 12 cols 192..207
    const float NEG = -__builtin_inff();
    #pragma unroll
    for (int v = 0; v < 8; ++v) {
        float m = NEG;
        #pragma unroll
        for (int t = 0; t < NTILES; ++t) {
            float val = (t < NTILES - 1 || lastValid) ? acc[t][v] : NEG;
            m = fmaxf(m, val);
        }
        #pragma unroll
        for (int d = 1; d < 16; d <<= 1) m = fmaxf(m, __shfl_xor(m, d, 32));

        float s = 0.0f;
        #pragma unroll
        for (int t = 0; t < NTILES; ++t) {
            if (t < NTILES - 1 || lastValid) s += __expf(acc[t][v] - m);
        }
        #pragma unroll
        for (int d = 1; d < 16; d <<= 1) s += __shfl_xor(s, d, 32);

        if (l15 == 0)
            conf[(size_t)b * SEQ + rowBase + v + half * 8] = 1.0f / s;
    }
}

// Kernel 2: per batch, stable descending rank of conf over S=1024 (matches
// jnp.argsort(-conf) tie-breaking), then gather the top-128 rows of x.
__global__ __launch_bounds__(1024)
void ws_select_gather_kernel(const float* __restrict__ x,
                             const float* __restrict__ conf,
                             float* __restrict__ out) {
    __shared__ int sel[NSEL];
    const int b = blockIdx.x;
    const int t = threadIdx.x;          // one thread per sequence position

    const float* cb = conf + (size_t)b * SEQ;
    const float  ci = cb[t];
    int rank = 0;
    for (int jj = 0; jj < SEQ; ++jj) {
        const float cj = cb[jj];
        rank += (cj > ci) || (cj == ci && jj < t);
    }
    if (rank < NSEL) sel[rank] = t;
    __syncthreads();

    // Gather 128 rows x 1536 floats with float4 moves.
    const int C4 = CH / 4;              // 384
    const float4* xb = (const float4*)(x + (size_t)b * SEQ * CH);
    float4*       ob = (float4*)(out + (size_t)b * NSEL * CH);
    for (int v = t; v < NSEL * C4; v += 1024) {
        const int row = v / C4;
        const int cw  = v - row * C4;
        ob[(size_t)row * C4 + cw] = xb[(size_t)sel[row] * C4 + cw];
    }
}

extern "C" void kernel_launch(void* const* d_in, const int* in_sizes, int n_in,
                              void* d_out, int out_size, void* d_ws, size_t ws_size,
                              hipStream_t stream) {
    (void)in_sizes; (void)n_in; (void)out_size; (void)ws_size;
    const float* x    = (const float*)d_in[0];   // (32,1024,1536) fp32
    const float* W    = (const float*)d_in[1];   // (200,1536)     fp32
    const float* bias = (const float*)d_in[2];   // (200,)         fp32
    // d_in[3] = num_select scalar (128), baked in as NSEL.

    float* conf = (float*)d_ws;                  // (32,1024) fp32 scratch
    float* out  = (float*)d_out;                 // (32,128,1536) fp32

    ws_gemm_conf_kernel<<<dim3(BATCH * (SEQ / 128)), dim3(256), 0, stream>>>(x, W, bias, conf);
    ws_select_gather_kernel<<<dim3(BATCH), dim3(1024), 0, stream>>>(x, conf, out);
}
